// IAF_10161892622818
// MI455X (gfx1250) — matrix operand order repaired and checked by reading, not compiled, not run on hardware
//
#include <hip/hip_runtime.h>

// ---------------------------------------------------------------------------
// MAF/IAF inverse pass for MI455X (gfx1250, wave32, WMMA).
// D=32, H=256, NB=2, B=16384.
// One persistent kernel: each wave owns 16 batch rows (= WMMA M dim) and runs
// the full 2-block x 32-step recurrence privately (recurrence is independent
// across batch rows). Masked weights are pre-packed to f16 in WMMA B-fragment
// layout by a tiny pack kernel so the hot loop is pure b128 loads + v_wmma.
// ---------------------------------------------------------------------------

#define D_ 32
#define H_ 256
#define NB_ 2
#define WAVES_ 4
#define TPB_ (WAVES_ * 32)

typedef __attribute__((ext_vector_type(16))) _Float16     v16h;
typedef __attribute__((ext_vector_type(8)))  float        v8f;
typedef __attribute__((ext_vector_type(4)))  unsigned int u32x4;

union FragU {
    u32x4        q[2];
    v16h         h;
    unsigned int u[8];
};

__device__ __forceinline__ unsigned int pack2h(float a, float b) {
    _Float16 ha = (_Float16)a;           // v_cvt_f16_f32 (RNE)
    _Float16 hb = (_Float16)b;
    unsigned short ua = __builtin_bit_cast(unsigned short, ha);
    unsigned short ub = __builtin_bit_cast(unsigned short, hb);
    return (unsigned int)ua | ((unsigned int)ub << 16);
}

// ---------------------------------------------------------------------------
// Pack kernel: build masked f16 weights in WMMA B-fragment order in d_ws.
//   B-fragment layout (16-bit B, KxN=K x 16), contiguous k order used here:
//     dword j (j=0..7), lane<16 : {B[2j][lane],   B[2j+1][lane]}
//                       lane>=16: {B[16+2j][l-16],B[17+2j][l-16]}
//   W0B: [blk][nTile(16)][lane(32)][8 dwords]      (B = (mask0*W0)^T, K=32)
//   W1B: [blk][nTile(16)][kChunk(8)][lane(32)][8]  (B = (mask1*W1)^T chunks)
// ---------------------------------------------------------------------------
#define W0B_DW (NB_ * 16 * 32 * 8)           // 8192 dwords  (32 KB)
#define W1B_DW (NB_ * 16 * 8 * 32 * 8)       // 65536 dwords (256 KB)

__global__ void iaf_pack(const float* __restrict__ W0,
                         const float* __restrict__ W1,
                         unsigned int* __restrict__ ws) {
    int t = blockIdx.x * blockDim.x + threadIdx.x;
    if (t < W0B_DW) {
        int j    = t & 7;
        int lane = (t >> 3) & 31;
        int nt   = (t >> 8) & 15;
        int blk  = t >> 12;
        int ke   = 2 * j + ((lane & 16) ? 16 : 0);       // even K index
        int n    = nt * 16 + (lane & 15);                // hidden unit (row of W0)
        int dn   = n % (D_ - 1);                         // m0 degree
        float v0 = (dn >= ke)     ? W0[(blk * H_ + n) * D_ + ke]     : 0.f;
        float v1 = (dn >= ke + 1) ? W0[(blk * H_ + n) * D_ + ke + 1] : 0.f;
        ws[t] = pack2h(v0, v1);
    } else if (t < W0B_DW + W1B_DW) {
        int t2   = t - W0B_DW;
        int j    = t2 & 7;
        int lane = (t2 >> 3) & 31;
        int kc   = (t2 >> 8) & 7;
        int nt   = (t2 >> 11) & 15;
        int blk  = t2 >> 15;
        int kl   = 2 * j + ((lane & 16) ? 16 : 0);
        int k    = kc * 32 + kl;                         // input hidden unit
        int n    = nt * 16 + (lane & 15);                // output hidden unit
        int dn   = n % (D_ - 1);
        int dk0  = k % (D_ - 1);
        int dk1  = (k + 1) % (D_ - 1);
        float v0 = (dn >= dk0) ? W1[(blk * H_ + n) * H_ + k]     : 0.f;
        float v1 = (dn >= dk1) ? W1[(blk * H_ + n) * H_ + k + 1] : 0.f;
        ws[t] = pack2h(v0, v1);
    }
}

// ---------------------------------------------------------------------------
// Main kernel: the entire flow in one launch. 4 waves/WG, 16 rows/wave.
// ---------------------------------------------------------------------------
__global__ __launch_bounds__(TPB_) void iaf_main(
        const float* __restrict__ x_in,
        const float* __restrict__ b0g, const float* __restrict__ b1g,
        const float* __restrict__ W2g, const float* __restrict__ b2g,
        const unsigned int* __restrict__ w0b,
        const unsigned int* __restrict__ w1b,
        float* __restrict__ out, int B) {

    __shared__ float ycur[WAVES_][16][D_];                                // 8 KB
    __shared__ float xcur[WAVES_][16][D_];                                // 8 KB
    __shared__ _Float16 __attribute__((aligned(16))) h0s[WAVES_][16][H_]; // 32 KB

    const int tid  = threadIdx.x;
    const int wave = tid >> 5;
    const int lane = tid & 31;
    const int lh   = lane >> 4;      // half-wave: 0 or 1
    const int ln   = lane & 15;
    const int row0 = (blockIdx.x * WAVES_ + wave) * 16;

    // Load target x tile (coalesced)
    for (int idx = lane; idx < 16 * D_; idx += 32) {
        int r = idx >> 5, c = idx & 31;
        xcur[wave][r][c] = x_in[(row0 + r) * D_ + c];
    }
    float ldaj = 0.f;

    for (int blk = 0; blk < NB_; ++blk) {
        for (int idx = lane; idx < 16 * D_; idx += 32) {
            int r = idx >> 5, c = idx & 31;
            ycur[wave][r][c] = 0.f;
        }
        __syncthreads();

        const unsigned int* w0blk = w0b + blk * (16 * 32 * 8);
        const unsigned int* w1blk = w1b + blk * (16 * 8 * 32 * 8);
        const float* b0blk = b0g + blk * H_;
        const float* b1blk = b1g + blk * H_;
        const float* b2blk = b2g + blk * 2 * D_;
        const float* W2blk = W2g + blk * 2 * D_ * H_;

        for (int i = 0; i < D_; ++i) {
            // ---- A-fragment for GEMM0: y tile (16x32 f16), contiguous k order ----
            FragU a0;
#pragma unroll
            for (int j = 0; j < 8; ++j) {
                int kb = (lh << 4) + (j << 1);
                a0.h[2 * j]     = (_Float16)ycur[wave][ln][kb];
                a0.h[2 * j + 1] = (_Float16)ycur[wave][ln][kb + 1];
            }
            // ---- GEMM0: h0 = relu(y @ (m0*W0)^T + b0), K = 32 = D ----
            for (int nt = 0; nt < 16; ++nt) {
                FragU b;
                const u32x4* p = (const u32x4*)(w0blk + (nt * 32 + lane) * 8);
                b.q[0] = p[0];
                b.q[1] = p[1];
                float bias = b0blk[nt * 16 + ln];
                v8f c;
#pragma unroll
                for (int v = 0; v < 8; ++v) c[v] = bias;
                c = __builtin_amdgcn_wmma_f32_16x16x32_f16(
                        false, a0.h, false, b.h, (short)0, c, false, false);
#pragma unroll
                for (int v = 0; v < 8; ++v) {
                    float r = c[v] > 0.f ? c[v] : 0.f;
                    h0s[wave][v + lh * 8][nt * 16 + ln] = (_Float16)r;  // C layout -> row major
                }
            }
            __syncthreads();

            // ---- A-fragments for GEMM1 from LDS h0 (two ds_load_b128 each) ----
            v16h a1[8];
#pragma unroll
            for (int k = 0; k < 8; ++k) {
                const u32x4* p = (const u32x4*)&h0s[wave][ln][k * 32 + lh * 16];
                FragU t;
                t.q[0] = p[0];      // K = base .. base+7
                t.q[1] = p[1];      // K = base+8 .. base+15 (+16 bytes)
                a1[k] = t.h;
            }

            // ---- GEMM1 (h1 = relu(h0 @ (m1*W1)^T + b1)) fused with layer 3 ----
            float pm[8], ps[8];
#pragma unroll
            for (int v = 0; v < 8; ++v) { pm[v] = 0.f; ps[v] = 0.f; }

            for (int nt = 0; nt < 16; ++nt) {
                const unsigned int* base = w1blk + (nt * 8) * 32 * 8;
                if (nt < 15)
                    __builtin_prefetch((const void*)(base + 8 * 32 * 8 + lane * 8), 0, 1);
                float bias = b1blk[nt * 16 + ln];
                v8f c;
#pragma unroll
                for (int v = 0; v < 8; ++v) c[v] = bias;
#pragma unroll
                for (int k = 0; k < 8; ++k) {
                    FragU b;
                    const u32x4* p = (const u32x4*)(base + (k * 32 + lane) * 8);
                    b.q[0] = p[0];
                    b.q[1] = p[1];
                    c = __builtin_amdgcn_wmma_f32_16x16x32_f16(
                            false, a1[k], false, b.h, (short)0, c, false, false);
                }
                // fused layer 3: only columns i and D+i are ever read at step i
                int hq   = nt * 16 + ln;
                int degh = hq % (D_ - 1);
                float wm = (degh < i) ? W2blk[i * H_ + hq]        : 0.f;
                float wv = (degh < i) ? W2blk[(D_ + i) * H_ + hq] : 0.f;
#pragma unroll
                for (int v = 0; v < 8; ++v) {
                    float r = c[v] > 0.f ? c[v] : 0.f;   // relu(h1)
                    pm[v] = __builtin_fmaf(r, wm, pm[v]);
                    ps[v] = __builtin_fmaf(r, wv, ps[v]);
                }
            }
            // reduce over the 16 lanes of each half-wave (N dimension)
#pragma unroll
            for (int off = 8; off >= 1; off >>= 1) {
#pragma unroll
                for (int v = 0; v < 8; ++v) {
                    pm[v] += __shfl_xor(pm[v], off, 32);
                    ps[v] += __shfl_xor(ps[v], off, 32);
                }
            }
            float bm = b2blk[i], bs = b2blk[D_ + i];
#pragma unroll
            for (int v = 0; v < 8; ++v) {
                if (ln == v) {                    // one owner lane per row per half
                    int r   = v + lh * 8;
                    float m = bm + pm[v];
                    float s = bs + ps[v];
                    ycur[wave][r][i] = (xcur[wave][r][i] - m) * __expf(-s);
                    ldaj -= s;
                }
            }
            __syncthreads();
        } // step i

        // x = reverse(y) for next block / final output
        for (int idx = lane; idx < 16 * D_; idx += 32) {
            int r = idx >> 5, c = idx & 31;
            xcur[wave][r][c] = ycur[wave][r][(D_ - 1) - c];
        }
        __syncthreads();
    } // blk

    // outputs: [B*D floats of x][B floats of log|det J|]
    for (int idx = lane; idx < 16 * D_; idx += 32) {
        int r = idx >> 5, c = idx & 31;
        out[(row0 + r) * D_ + c] = xcur[wave][r][c];
    }
#pragma unroll
    for (int v = 0; v < 8; ++v)
        if (ln == v) out[B * D_ + row0 + v + lh * 8] = ldaj;
}

// ---------------------------------------------------------------------------
extern "C" void kernel_launch(void* const* d_in, const int* in_sizes, int n_in,
                              void* d_out, int out_size, void* d_ws, size_t ws_size,
                              hipStream_t stream) {
    const float* x  = (const float*)d_in[0];
    const float* W0 = (const float*)d_in[1];
    const float* b0 = (const float*)d_in[2];
    const float* W1 = (const float*)d_in[3];
    const float* b1 = (const float*)d_in[4];
    const float* W2 = (const float*)d_in[5];
    const float* b2 = (const float*)d_in[6];
    const int B = in_sizes[0] / D_;

    unsigned int* ws  = (unsigned int*)d_ws;
    unsigned int* w0b = ws;
    unsigned int* w1b = ws + W0B_DW;

    const int packN = W0B_DW + W1B_DW;
    iaf_pack<<<(packN + 255) / 256, 256, 0, stream>>>(W0, W1, ws);

    const int rowsPerWG = WAVES_ * 16;
    iaf_main<<<B / rowsPerWG, TPB_, 0, stream>>>(x, b0, b1, W2, b2, w0b, w1b,
                                                 (float*)d_out, B);
}
